// NaiveGroupConv2d_62740882260490
// MI455X (gfx1250) — matrix operand order repaired and checked by reading, not compiled, not run on hardware
//
#include <hip/hip_runtime.h>

#define BATCH 32
#define HW 56
#define CH 256
#define PAIR_CH 16      // two 8-channel groups fused per WMMA tile (N=16)
#define NPAIR 16
#define HPAIRS 28       // 56 rows / 2 rows per block
#define NVALID (BATCH * HW * HW)   // 100352 pixels per channel
#define EPS 1e-5f

#define XWP 64          // staged width (padded pow2): w = -1 .. 62 (>=56 zero)
#define XROWS 4         // input rows h0-1 .. h0+2

typedef __attribute__((ext_vector_type(16))) __bf16 v16bf;
typedef __attribute__((ext_vector_type(8)))  __bf16 v8bf;
typedef __attribute__((ext_vector_type(8)))  float  v8f;

// ---------------------------------------------------------------------------
// Kernel 1: grouped conv via bf16 WMMA (f32 accum) + bias + BN partial sums.
// One wave = 16 pixels (along W) x 16 out-channels (a pair of groups).
// K dim: kidx = tap*16 + channel_in_pair; taps 0..8 real, tap 9 zero-pad
// => 5 x v_wmma_f32_16x16x32_bf16 per tile with block-diagonal B.
// All operands staged in LDS as bf16 once per block; the MMA loop is pure
// ds_load_b128 + v_wmma.
// ---------------------------------------------------------------------------
__global__ void __launch_bounds__(256)
conv_wmma_kernel(const float* __restrict__ x, const float* __restrict__ wgt,
                 const float* __restrict__ bias, float* __restrict__ y,
                 float* __restrict__ gsum, float* __restrict__ gsumsq) {
  // x tile: [row(4)][half(2)][wloc(64)][ci(8)] bf16 = 8 KB (shift-only idx)
  __shared__ __align__(16) __bf16 s_x[XROWS * 2 * XWP * 8];
  // block-diagonal pair weights incl. explicit zero tap 9:
  // [tap(10)][n(16)][cig(16)] bf16 = 5 KB
  __shared__ __align__(16) __bf16 s_wbd[10 * PAIR_CH * PAIR_CH];
  __shared__ float s_red[32];              // per-block channel sum/sumsq

  const int tid  = threadIdx.x;
  const int hp   = blockIdx.x;             // 3-D grid: no div/mod chains
  const int bb   = blockIdx.y;
  const int pair = blockIdx.z;
  const int cobase = pair * PAIR_CH;
  const int h0   = hp * 2;

  if (tid < 32) s_red[tid] = 0.f;

  // ---- stage x rows h0-1..h0+2, w=-1..62 (zero beyond 55), as bf16 ----
  // 512 chunks of 8 channels -> exactly 2 iterations, pow2 decomposition.
  for (int c = tid; c < XROWS * 2 * XWP; c += 256) {
    const int row  = c >> 7;
    const int hf   = (c >> 6) & 1;
    const int wloc = c & 63;
    const int hin  = h0 + row - 1;
    const int win  = wloc - 1;
    float4 lo = make_float4(0.f, 0.f, 0.f, 0.f);
    float4 hi4 = lo;
    if (((unsigned)hin < HW) & ((unsigned)win < HW)) {
      const float4* p4 = (const float4*)(
          x + (((size_t)(bb * HW + hin) * HW + win) * CH + pair * PAIR_CH + hf * 8));
      lo = p4[0]; hi4 = p4[1];
    }
    v8bf pk;
    pk[0] = (__bf16)lo.x;  pk[1] = (__bf16)lo.y;
    pk[2] = (__bf16)lo.z;  pk[3] = (__bf16)lo.w;
    pk[4] = (__bf16)hi4.x; pk[5] = (__bf16)hi4.y;
    pk[6] = (__bf16)hi4.z; pk[7] = (__bf16)hi4.w;
    *(v8bf*)&s_x[c * 8] = pk;
  }

  // ---- stage block-diagonal weights (zeros placed here, incl. tap 9) ----
  for (int i = tid; i < 10 * PAIR_CH * PAIR_CH; i += 256) {
    const int tap = i >> 8;
    const int rem = i & 255;
    const int n   = rem >> 4;
    const int cig = rem & 15;
    float v = 0.f;
    if ((tap < 9) && ((cig >> 3) == (n >> 3)))
      v = wgt[(tap * 8 + (cig & 7)) * CH + cobase + n];
    s_wbd[i] = (__bf16)v;
  }
  __syncthreads();

  const int wv    = tid >> 5;             // 8 waves per block
  const int lane  = tid & 31;
  const int wv4   = wv >> 2;              // which of the 2 output rows
  const int h     = h0 + wv4;
  const int wtile = wv & 3;               // 4 x 16-pixel tiles cover W=56 (+pad)
  const int w0    = wtile * 16;

  const int m  = lane & 15;               // A row / B col / C col
  const int hi = lane >> 4;               // lane half selects K sub-range
  const int wpix = w0 + m;

  v8bf zrow;
#pragma unroll
  for (int e = 0; e < 8; ++e) zrow[e] = (__bf16)0.0f;

  v8f acc = {};

#pragma unroll
  for (int t = 0; t < 5; ++t) {
    // ---- A fragment (ISA 16-bit A 16x32 layout), one ds_load_b128 per half ---
    // lane<16 : elems 0..7 -> tap 2t, cig 0..7 ; elems 8..15 -> tap 2t+1
    // lane>=16: same taps, cig 8..15   (tap = 2t+half is compile-time)
    v8bf ah[2];
#pragma unroll
    for (int half = 0; half < 2; ++half) {
      const int tap = 2 * t + half;
      if (tap < 9) {                       // folds at compile time
        const int kh = tap / 3, kw = tap % 3;
        int wp1 = wpix + kw;               // = win + 1, >= 0, <= 65
        wp1 = (wp1 > XWP - 1) ? (XWP - 1) : wp1; // clamp pad pixels (discarded)
        const int rowhi = (((wv4 + kh) << 1) | hi);
        ah[half] = *(const v8bf*)&s_x[((rowhi << 6) + wp1) * 8];
      } else {
        ah[half] = zrow;                   // zero-pad tap 9 (t==4 only)
      }
    }
    const v16bf a = __builtin_shufflevector(ah[0], ah[1],
        0, 1, 2, 3, 4, 5, 6, 7, 8, 9, 10, 11, 12, 13, 14, 15);

    // ---- B fragment: straight 32B LDS read, zeros pre-placed ----
    // elems 0..7 -> tap 2t, cig 8*hi..+7 ; elems 8..15 -> tap 2t+1, cig 8*hi..+7
    const int tap_b = 2 * t;               // 0..9, tap 9 row is zeros
    const v8bf b0 = *(const v8bf*)&s_wbd[(tap_b * PAIR_CH + m) * PAIR_CH + hi * 8];
    const v8bf b1 = *(const v8bf*)&s_wbd[((tap_b + 1) * PAIR_CH + m) * PAIR_CH + hi * 8];
    const v16bf bm = __builtin_shufflevector(b0, b1,
        0, 1, 2, 3, 4, 5, 6, 7, 8, 9, 10, 11, 12, 13, 14, 15);

    acc = __builtin_amdgcn_wmma_f32_16x16x32_bf16(false, a, false, bm,
                                                  (short)0, acc, false, false);
  }

  // ---- epilogue: bias, store y (pre-BN), BN partial sums ----
  const int co = cobase + m;              // C/D: lane holds column N = lane%16
  const float bv = bias[co];
  float psum = 0.f, psq = 0.f;
#pragma unroll
  for (int r = 0; r < 8; ++r) {
    const int M  = r + hi * 8;            // C/D: VGPR r -> row M (+8 for hi lanes)
    const int wp = w0 + M;
    const float yv = acc[r] + bv;
    if (wp < HW) {
      y[(((size_t)(bb * HW + h) * HW + wp) * CH) + co] = yv;
      psum += yv;
      psq  += yv * yv;
    }
  }
  atomicAdd(&s_red[m], psum);
  atomicAdd(&s_red[16 + m], psq);
  __syncthreads();
  if (tid < 16)       atomicAdd(&gsum[cobase + tid], s_red[tid]);
  else if (tid < 32)  atomicAdd(&gsumsq[cobase + tid - 16], s_red[tid]);
}

// ---------------------------------------------------------------------------
// Kernel 0: zero the 512-float stats region (d_ws is poisoned by the harness).
// ---------------------------------------------------------------------------
__global__ void init_stats_kernel(float* ws) {
  const int i = blockIdx.x * blockDim.x + threadIdx.x;
  if (i < 512) ws[i] = 0.f;
}

// ---------------------------------------------------------------------------
// Kernel 2: fold mean/var/gamma/beta into per-channel scale & shift.
// ---------------------------------------------------------------------------
__global__ void finalize_stats_kernel(const float* __restrict__ gsum,
                                      const float* __restrict__ gsumsq,
                                      const float* __restrict__ gamma,
                                      const float* __restrict__ beta,
                                      float* __restrict__ scale,
                                      float* __restrict__ shift) {
  const int c = threadIdx.x;
  const float inv_n = 1.0f / (float)NVALID;
  const float mean = gsum[c] * inv_n;
  const float var  = gsumsq[c] * inv_n - mean * mean;
  const float rs   = rsqrtf(var + EPS);
  const float sc   = gamma[c] * rs;
  scale[c] = sc;
  shift[c] = beta[c] - mean * sc;
}

// ---------------------------------------------------------------------------
// Kernel 3: y = relu(y*scale[c] + shift[c]), in place, float4 vectorized.
// ---------------------------------------------------------------------------
__global__ void __launch_bounds__(256)
bn_relu_kernel(float* __restrict__ y, const float* __restrict__ scale,
               const float* __restrict__ shift) {
  const size_t idx = (size_t)blockIdx.x * 256 + threadIdx.x;  // float4 index
  const int c4 = (int)(idx & 63);                             // 64 float4 per pixel
  float4 v        = ((const float4*)y)[idx];
  const float4 sc = ((const float4*)scale)[c4];
  const float4 sh = ((const float4*)shift)[c4];
  v.x = fmaxf(fmaf(v.x, sc.x, sh.x), 0.f);
  v.y = fmaxf(fmaf(v.y, sc.y, sh.y), 0.f);
  v.z = fmaxf(fmaf(v.z, sc.z, sh.z), 0.f);
  v.w = fmaxf(fmaf(v.w, sc.w, sh.w), 0.f);
  ((float4*)y)[idx] = v;
}

extern "C" void kernel_launch(void* const* d_in, const int* in_sizes, int n_in,
                              void* d_out, int out_size, void* d_ws, size_t ws_size,
                              hipStream_t stream) {
  const float* x     = (const float*)d_in[0];
  const float* w     = (const float*)d_in[1];
  const float* bias  = (const float*)d_in[2];
  const float* gamma = (const float*)d_in[3];
  const float* beta  = (const float*)d_in[4];
  float* out = (float*)d_out;

  float* ws     = (float*)d_ws;
  float* gsum   = ws;          // [256]
  float* gsumsq = ws + 256;    // [256]
  float* scale  = ws + 512;    // [256]
  float* shift  = ws + 768;    // [256]

  init_stats_kernel<<<2, 256, 0, stream>>>(ws);

  // 3-D grid: x=row-pairs, y=batch, z=group-pairs; 8 waves/block
  conv_wmma_kernel<<<dim3(HPAIRS, BATCH, NPAIR), 256, 0, stream>>>(
      x, w, bias, out, gsum, gsumsq);

  finalize_stats_kernel<<<1, 256, 0, stream>>>(gsum, gsumsq, gamma, beta,
                                               scale, shift);

  // 32*56*56*256 / 4 = 6,422,528 float4 = 25088 blocks x 256 threads
  bn_relu_kernel<<<25088, 256, 0, stream>>>(out, scale, shift);
}
